// MMGCN_54176717472163
// MI455X (gfx1250) — matrix operand order repaired and checked by reading, not compiled, not run on hardware
//
#include <hip/hip_runtime.h>

typedef __attribute__((ext_vector_type(16))) _Float16 v16h;
typedef __attribute__((ext_vector_type(8)))  float    v8f;

#define NDIM 64
#define NEG_SLOPE 0.01f

static __device__ __forceinline__ v8f wmma16(v16h a, v16h b, v8f c) {
  // D = A(16x32 f16) * B(32x16 f16) + C(16x16 f32)
  return __builtin_amdgcn_wmma_f32_16x16x32_f16(
      /*neg_a=*/false, a, /*neg_b=*/false, b,
      /*c_mod=*/(short)0, c, /*reuse_a=*/false, /*reuse_b=*/false);
}

static __device__ __forceinline__ float leaky(float v) {
  return v > 0.0f ? v : NEG_SLOPE * v;
}

// ---------------------------------------------------------------------------
// Pre-pack one layer slice W[3][64][64] (f32, row-major) into WMMA B-operand
// layout as f16. Operand (g, jt, ch) for lane l lives at
//   dst[(((g*4 + jt)*2 + ch)*32 + l) * 16 .. +15]
// with b[i] = W[g][jt*16 + (l&15)][ch*32 + (l>>4)*8 + (i<8 ? i : i+8)].
// 768 threads per slice; each writes 16 contiguous halves (32B).
// ---------------------------------------------------------------------------
__global__ __launch_bounds__(256) void pack_weights_kernel(
    const float* __restrict__ W, _Float16* __restrict__ dst) {
  const int t = blockIdx.x * 256 + threadIdx.x;   // [0, 768)
  if (t >= 3 * 4 * 2 * 32) return;
  const int lane = t & 31;
  const int rest = t >> 5;          // [0, 24)
  const int ch   = rest & 1;
  const int jt   = (rest >> 1) & 3;
  const int g    = rest >> 3;       // [0, 3)
  const int n  = lane & 15;
  const int hi = lane >> 4;
  const float* __restrict__ src =
      W + g * NDIM * NDIM + (jt * 16 + n) * NDIM + ch * 32 + hi * 8;
  _Float16* __restrict__ d = dst + (size_t)t * 16;
#pragma unroll
  for (int i = 0; i < 16; ++i) d[i] = (_Float16)src[i < 8 ? i : i + 8];
}

// ---------------------------------------------------------------------------
// x = cat(pref, feat); x /= max(||x||_2, 1e-12)   (one wave per row)
// ---------------------------------------------------------------------------
__global__ __launch_bounds__(256) void normalize_kernel(
    const float* __restrict__ pref, int npref,
    const float* __restrict__ feat,
    float* __restrict__ x, int nrows) {
  const int wave = threadIdx.x >> 5;
  const int lane = threadIdx.x & 31;
  const int row  = blockIdx.x * 8 + wave;
  if (row >= nrows) return;
  const float* __restrict__ src =
      (row < npref) ? (pref + (size_t)row * NDIM)
                    : (feat + (size_t)(row - npref) * NDIM);
  float2 v = ((const float2*)src)[lane];
  float ss = v.x * v.x + v.y * v.y;
#pragma unroll
  for (int off = 16; off > 0; off >>= 1) ss += __shfl_xor(ss, off, 32);
  const float inv = 1.0f / fmaxf(sqrtf(ss), 1e-12f);
  float2 o;
  o.x = v.x * inv;
  o.y = v.y * inv;
  ((float2*)(x + (size_t)row * NDIM))[lane] = o;
}

__global__ __launch_bounds__(256) void zero_f32_kernel(float* __restrict__ p, int n) {
  const int i = blockIdx.x * blockDim.x + threadIdx.x;
  if (i < n) p[i] = 0.0f;
}

__global__ __launch_bounds__(256) void degree_kernel(
    const int* __restrict__ row, int ne, float* __restrict__ deg) {
  const int e = blockIdx.x * blockDim.x + threadIdx.x;
  if (e < ne) atomicAdd(&deg[row[e]], 1.0f);
}

// agg[row[e]][c] += x[col[e]][c]   (thread = (edge, component))
__global__ __launch_bounds__(256) void scatter_kernel(
    const int* __restrict__ row, const int* __restrict__ col, int ne,
    const float* __restrict__ x, float* __restrict__ agg) {
  const int idx = blockIdx.x * blockDim.x + threadIdx.x;
  const int e = idx >> 6;
  const int c = idx & 63;
  if (e < ne) {
    const int r = row[e];
    const int s = col[e];
    atomicAdd(&agg[(size_t)r * NDIM + c], x[(size_t)s * NDIM + c]);
  }
}

// ---------------------------------------------------------------------------
// One GCN layer, fused, in place:
//   h     = leaky((agg/deg) @ W0^T + b0)
//   x_hat = leaky(x @ W1^T + b1) + id_emb
//   x     = leaky(h @ W2^T + b2 + x_hat)
// One wave per 16-row node tile; weights pre-packed in B layout (f16).
// ---------------------------------------------------------------------------
__global__ __launch_bounds__(256) void mmgcn_layer_kernel(
    const float* __restrict__ agg,
    float* __restrict__ x,               // in/out [80000][64]
    const float* __restrict__ deg,       // [80000]
    const v16h*  __restrict__ Wp,        // packed [3][4][2][32] v16h (this layer)
    const float* __restrict__ bias,      // [3][64]
    const float* __restrict__ id_emb) {  // [80000][64]
  __shared__ float hbuf[8][16 * NDIM];   // 4KB/wave: C-layout -> A-layout relay

  const int wave = threadIdx.x >> 5;
  const int lane = threadIdx.x & 31;
  const int n    = lane & 15;            // A row (M) and B/C column (N)
  const int hi   = lane >> 4;            // lane half selects K / M sub-block
  const int row0 = (blockIdx.x * 8 + wave) * 16;

  // ---- A operands: agg tile (scaled by 1/deg) and x tile, f32 -> f16 ----
  const int arow = row0 + n;
  const float invd = 1.0f / deg[arow];
  const float* __restrict__ aggp = agg + (size_t)arow * NDIM;
  const float* __restrict__ xp   = x   + (size_t)arow * NDIM;

  v16h aA0, aA1, aX0, aX1;               // K chunks [0,32) and [32,64)
#pragma unroll
  for (int i = 0; i < 8; ++i) {
    const int k = hi * 8 + i;
    aA0[i]     = (_Float16)(aggp[k]      * invd);
    aA0[8 + i] = (_Float16)(aggp[16 + k] * invd);
    aA1[i]     = (_Float16)(aggp[32 + k] * invd);
    aA1[8 + i] = (_Float16)(aggp[48 + k] * invd);
    aX0[i]     = (_Float16)xp[k];
    aX0[8 + i] = (_Float16)xp[16 + k];
    aX1[i]     = (_Float16)xp[32 + k];
    aX1[8 + i] = (_Float16)xp[48 + k];
  }

  // packed B operand fetch: (g, jt, ch) -> one aligned 32B v16h per lane
#define BOP(g, jt, ch) Wp[(((g) * 4 + (jt)) * 2 + (ch)) * 32 + lane]

  // ---- x_hat = leaky(x @ W1^T + b1) + id ----
  v8f xh[4];
#pragma unroll
  for (int jt = 0; jt < 4; ++jt) {
    const int j = jt * 16 + n;
    const float b1v = bias[64 + j];
    v8f c;
#pragma unroll
    for (int r = 0; r < 8; ++r) c[r] = b1v;   // bias folded into C
    c = wmma16(aX0, BOP(1, jt, 0), c);
    c = wmma16(aX1, BOP(1, jt, 1), c);
#pragma unroll
    for (int r = 0; r < 8; ++r)
      c[r] = leaky(c[r]) + id_emb[(size_t)(row0 + hi * 8 + r) * NDIM + j];
    xh[jt] = c;
  }

  // ---- h = leaky(agg/deg @ W0^T + b0); relay via LDS to rebuild A layout ----
  float* __restrict__ hb = hbuf[wave];
#pragma unroll
  for (int jt = 0; jt < 4; ++jt) {
    const int j = jt * 16 + n;
    const float b0v = bias[j];
    v8f c;
#pragma unroll
    for (int r = 0; r < 8; ++r) c[r] = b0v;
    c = wmma16(aA0, BOP(0, jt, 0), c);
    c = wmma16(aA1, BOP(0, jt, 1), c);
#pragma unroll
    for (int r = 0; r < 8; ++r)
      hb[(hi * 8 + r) * NDIM + j] = leaky(c[r]);  // C layout -> row-major LDS
  }
  // same-wave LDS ops are in-order; private slab per wave -> no barrier needed
  v16h hA0, hA1;
  const float* __restrict__ hrow = hb + n * NDIM;
#pragma unroll
  for (int i = 0; i < 8; ++i) {
    const int k = hi * 8 + i;
    hA0[i]     = (_Float16)hrow[k];
    hA0[8 + i] = (_Float16)hrow[16 + k];
    hA1[i]     = (_Float16)hrow[32 + k];
    hA1[8 + i] = (_Float16)hrow[48 + k];
  }

  // ---- x = leaky(h @ W2^T + b2 + x_hat)  (in-place: tile-local rows only) ----
#pragma unroll
  for (int jt = 0; jt < 4; ++jt) {
    const int j = jt * 16 + n;
    const float b2v = bias[128 + j];
    v8f c = xh[jt];
#pragma unroll
    for (int r = 0; r < 8; ++r) c[r] += b2v;
    c = wmma16(hA0, BOP(2, jt, 0), c);
    c = wmma16(hA1, BOP(2, jt, 1), c);
#pragma unroll
    for (int r = 0; r < 8; ++r)
      x[(size_t)(row0 + hi * 8 + r) * NDIM + j] = leaky(c[r]);
  }
#undef BOP
}

__global__ __launch_bounds__(256) void combine_kernel(
    const float* __restrict__ a, const float* __restrict__ b,
    float* __restrict__ out, int n) {
  const int i = blockIdx.x * blockDim.x + threadIdx.x;
  if (i < n) out[i] = 0.5f * (a[i] + b[i]);
}

extern "C" void kernel_launch(void* const* d_in, const int* in_sizes, int n_in,
                              void* d_out, int out_size, void* d_ws, size_t ws_size,
                              hipStream_t stream) {
  const float* v_feat = (const float*)d_in[0];  // [30000,64]
  const float* t_feat = (const float*)d_in[1];  // [30000,64]
  const float* id_emb = (const float*)d_in[2];  // [80000,64]
  const float* v_pref = (const float*)d_in[3];  // [50000,64]
  const float* t_pref = (const float*)d_in[4];  // [50000,64]
  const float* v_W    = (const float*)d_in[5];  // [2,3,64,64]
  const float* v_b    = (const float*)d_in[6];  // [2,3,64]
  const float* t_W    = (const float*)d_in[7];
  const float* t_b    = (const float*)d_in[8];
  const int*   edge   = (const int*)d_in[9];    // [2, NE]

  const int NU = 50000;
  const int NN = 80000;
  const int NE = in_sizes[9] / 2;
  const int* erow = edge;
  const int* ecol = edge + NE;

  float* xv  = (float*)d_ws;                  // [80000*64] f32
  float* xt  = xv  + (size_t)NN * NDIM;       // [80000*64] f32
  float* agg = xt  + (size_t)NN * NDIM;       // [80000*64] f32
  float* deg = agg + (size_t)NN * NDIM;       // [80000]    f32
  _Float16* wpk = (_Float16*)(deg + NN);      // 4 slices x 768 x 16 halves (96KB)
  const size_t SLICE = 768 * 16;              // halves per packed layer slice
  (void)ws_size; (void)n_in;

  const int NVEC = NN * NDIM;                 // 5,120,000

  // pack all 4 weight slices (v-layer0, v-layer1, t-layer0, t-layer1)
  for (int layer = 0; layer < 2; ++layer) {
    pack_weights_kernel<<<3, 256, 0, stream>>>(
        v_W + layer * 3 * NDIM * NDIM, wpk + (size_t)layer * SLICE);
    pack_weights_kernel<<<3, 256, 0, stream>>>(
        t_W + layer * 3 * NDIM * NDIM, wpk + (size_t)(2 + layer) * SLICE);
  }

  // shared degree (bincount of row), as float divisor
  zero_f32_kernel<<<(NN + 255) / 256, 256, 0, stream>>>(deg, NN);
  degree_kernel<<<(NE + 255) / 256, 256, 0, stream>>>(erow, NE, deg);

  // normalized inputs for both towers
  normalize_kernel<<<NN / 8, 256, 0, stream>>>(v_pref, NU, v_feat, xv, NN);
  normalize_kernel<<<NN / 8, 256, 0, stream>>>(t_pref, NU, t_feat, xt, NN);

  const int tiles_grid = NN / (16 * 8);       // 625 blocks, 8 waves * 16 rows each
  for (int layer = 0; layer < 2; ++layer) {
    // tower v
    zero_f32_kernel<<<(NVEC + 255) / 256, 256, 0, stream>>>(agg, NVEC);
    scatter_kernel<<<(NE * 64 + 255) / 256, 256, 0, stream>>>(erow, ecol, NE, xv, agg);
    mmgcn_layer_kernel<<<tiles_grid, 256, 0, stream>>>(
        agg, xv, deg, (const v16h*)(wpk + (size_t)layer * SLICE),
        v_b + layer * 3 * NDIM, id_emb);
    // tower t
    zero_f32_kernel<<<(NVEC + 255) / 256, 256, 0, stream>>>(agg, NVEC);
    scatter_kernel<<<(NE * 64 + 255) / 256, 256, 0, stream>>>(erow, ecol, NE, xt, agg);
    mmgcn_layer_kernel<<<tiles_grid, 256, 0, stream>>>(
        agg, xt, deg, (const v16h*)(wpk + (size_t)(2 + layer) * SLICE),
        t_b + layer * 3 * NDIM, id_emb);
  }

  combine_kernel<<<(out_size + 255) / 256, 256, 0, stream>>>(xv, xt, (float*)d_out, out_size);
}
